// GCNConv_17497696764534
// MI455X (gfx1250) — compile-verified, matrix operand order checked by reading, not run on hardware
//
#include <hip/hip_runtime.h>
#include <hip/hip_bf16.h>

typedef __attribute__((ext_vector_type(16))) __bf16 v16bf;
typedef __attribute__((ext_vector_type(8)))  __bf16 v8bf;
typedef __attribute__((ext_vector_type(8)))  float  v8f;

#define DFEAT 128

// round-to-nearest-even fp32 -> bf16 (bit-level; used only in memory-bound pre-pass)
__device__ __forceinline__ __bf16 to_bf16(float f) {
    union { float f; unsigned u; } v; v.f = f;
    unsigned r = (v.u + 0x7FFFu + ((v.u >> 16) & 1u)) >> 16;
    union { unsigned short s; __bf16 b; } o; o.s = (unsigned short)r;
    return o.b;
}

// ---------------- degree / normalization ----------------

__global__ void deg_init_kernel(float* __restrict__ node_deg, int n) {
    int i = blockIdx.x * blockDim.x + threadIdx.x;
    if (i < n) node_deg[i] = 1.0f;   // self-loop weight 1.0
}

__global__ void deg_accum_kernel(const int* __restrict__ rows,
                                 const float* __restrict__ ew,
                                 float* __restrict__ node_deg, int nE) {
    int e = blockIdx.x * blockDim.x + threadIdx.x;
    if (e < nE) atomicAdd(node_deg + rows[e], ew[e]);
}

__global__ void dinv_kernel(const float* __restrict__ node_deg,
                            float* __restrict__ dinv, int n) {
    int i = blockIdx.x * blockDim.x + threadIdx.x;
    if (i < n) {
        float d = node_deg[i];
        dinv[i] = (d > 0.0f) ? rsqrtf(d) : 0.0f;
    }
}

// ---------------- fp32 -> bf16 pre-pass (vectorized x4) ----------------

__global__ void cvt_bf16_kernel(const float* __restrict__ in,
                                __bf16* __restrict__ out, long long n4) {
    long long i = (long long)blockIdx.x * blockDim.x + threadIdx.x;
    if (i >= n4) return;
    const float4 f = ((const float4*)in)[i];
    v8bf* o8 = (v8bf*)out;               // write 4 bf16 = 8 bytes
    __bf16 tmp[4] = { to_bf16(f.x), to_bf16(f.y), to_bf16(f.z), to_bf16(f.w) };
    ((ulonglong1*)((char*)out + i * 8))->x = *(const unsigned long long*)tmp;
    (void)o8;
}

// ---------------- GEMM: y = x @ W^T + b  (bf16 WMMA, f32 accumulate) ----------
// Block = 256 threads = 8 waves; block covers 64 rows (4 M-tiles of 16).
// Wave w owns N-columns [16w,16w+16) and keeps its full K=128 B-fragments in
// registers (4 x v16bf). Per M-tile: 4 x v_wmma_f32_16x16x32_bf16.
// Fragment layout (ISA 7.12.2, wave32): lane = 16*half + l15.
//   A: row = l15; elements e<8 -> k = koff + 8*half + e (contiguous!)
//                 e>=8 -> k = koff + 16 + 8*half + (e-8) (contiguous!)
//   -> each half-fragment is one aligned 16-byte (8 x bf16) load.
//   B[k][n] = W[n][k]: same k map, n = l15, contiguous along W's row.
//   D: VGPR r -> row (r + 8*half), col l15.

__global__ __launch_bounds__(256) void gemm_bias_wmma_kernel(
    const __bf16* __restrict__ xb,   // [M,128] bf16 row-major
    const __bf16* __restrict__ Wb,   // [128,128] bf16 row-major
    const float*  __restrict__ bias, // [128]
    float* __restrict__ y,           // [M,128]
    int M)
{
    const int mbase = blockIdx.x * 64;
    const int wave  = threadIdx.x >> 5;
    const int lane  = threadIdx.x & 31;
    const int half  = lane >> 4;
    const int l15   = lane & 15;
    const int ncol  = (wave << 4) + l15;

    // B fragments for all 4 K-chunks (held in VGPRs for the whole block)
    const __bf16* __restrict__ bp = Wb + (size_t)ncol * DFEAT;
    v16bf bfrag[4];
    #pragma unroll
    for (int c = 0; c < 4; ++c) {
        v8bf b0 = *(const v8bf*)(bp + 32 * c + 8 * half);
        v8bf b1 = *(const v8bf*)(bp + 32 * c + 16 + 8 * half);
        bfrag[c] = __builtin_shufflevector(b0, b1,
                    0, 1, 2, 3, 4, 5, 6, 7, 8, 9, 10, 11, 12, 13, 14, 15);
    }
    const float bv = bias[ncol];

    #pragma unroll
    for (int mt = 0; mt < 4; ++mt) {
        int arow = mbase + mt * 16 + l15;
        if (arow >= M) arow = M - 1;           // clamp loads; stores are guarded
        const __bf16* __restrict__ ap = xb + (size_t)arow * DFEAT;

        v8f acc = {};
        #pragma unroll
        for (int c = 0; c < 4; ++c) {
            v8bf a0 = *(const v8bf*)(ap + 32 * c + 8 * half);
            v8bf a1 = *(const v8bf*)(ap + 32 * c + 16 + 8 * half);
            v16bf afrag = __builtin_shufflevector(a0, a1,
                    0, 1, 2, 3, 4, 5, 6, 7, 8, 9, 10, 11, 12, 13, 14, 15);
            acc = __builtin_amdgcn_wmma_f32_16x16x32_bf16(
                /*neg_a=*/false, afrag, /*neg_b=*/false, bfrag[c],
                /*c_mod=*/(short)0, acc, /*reuse_a=*/false, /*reuse_b=*/false);
        }

        #pragma unroll
        for (int r = 0; r < 8; ++r) {
            const int mrow = mbase + mt * 16 + r + 8 * half;
            if (mrow < M) y[(size_t)mrow * DFEAT + ncol] = acc[r] + bv;
        }
    }
}

// ---------------- SpMM aggregation ----------------

__global__ void coef_kernel(const int* __restrict__ rows,
                            const int* __restrict__ cols,
                            const float* __restrict__ ew,
                            const float* __restrict__ dinv,
                            float* __restrict__ coef, int nE) {
    int e = blockIdx.x * blockDim.x + threadIdx.x;
    if (e < nE) coef[e] = dinv[rows[e]] * ew[e] * dinv[cols[e]];
}

// out[i] = dinv[i]^2 * y[i]   (self-loop term; also initializes out)
__global__ void selfloop_kernel(const float* __restrict__ y,
                                const float* __restrict__ dinv,
                                float* __restrict__ out, int n) {
    long long t = (long long)blockIdx.x * blockDim.x + threadIdx.x;
    long long total = (long long)n * DFEAT;
    if (t >= total) return;
    int i = (int)(t >> 7);
    float di = dinv[i];
    out[t] = di * di * y[t];
}

// out[row[e]][f] += coef[e] * y[col[e]][f]
__global__ void edge_scatter_kernel(const int* __restrict__ rows,
                                    const int* __restrict__ cols,
                                    const float* __restrict__ coef,
                                    const float* __restrict__ y,
                                    float* __restrict__ out, int nE) {
    long long t = (long long)blockIdx.x * blockDim.x + threadIdx.x;
    long long total = (long long)nE * DFEAT;
    if (t >= total) return;
    int e = (int)(t >> 7);
    int f = (int)(t & (DFEAT - 1));
    int r = rows[e];
    int c = cols[e];
    atomicAdd(out + (size_t)r * DFEAT + f, coef[e] * y[(size_t)c * DFEAT + f]);
}

__global__ void relu_kernel(float* __restrict__ out, long long total) {
    long long t = (long long)blockIdx.x * blockDim.x + threadIdx.x;
    if (t < total) out[t] = fmaxf(out[t], 0.0f);
}

// ---------------- launch ----------------

static inline size_t align_up(size_t v, size_t a) { return (v + a - 1) & ~(a - 1); }

extern "C" void kernel_launch(void* const* d_in, const int* in_sizes, int n_in,
                              void* d_out, int out_size, void* d_ws, size_t ws_size,
                              hipStream_t stream) {
    const float* x  = (const float*)d_in[0];   // [N,128]
    const float* W  = (const float*)d_in[1];   // [128,128]
    const float* b  = (const float*)d_in[2];   // [128]
    const float* ew = (const float*)d_in[3];   // [E]
    const int*   ei = (const int*)d_in[4];     // [2,E] flat (int32 per harness)

    const int N = in_sizes[0] / DFEAT;         // 50000
    const int E = in_sizes[3];                 // 800000
    const int* rows = ei;
    const int* cols = ei + E;

    float* out = (float*)d_out;

    // workspace layout (each region 256B-aligned):
    //   node_deg[N] f32 | dinv[N] f32 | coef[E] f32 | xb[N*128] bf16 |
    //   Wb[128*128] bf16 | y[N*128] f32
    char* ws = (char*)d_ws;
    size_t off = 0;
    float* node_deg = (float*)(ws + off);  off = align_up(off + (size_t)N * 4, 256);
    float* dinv     = (float*)(ws + off);  off = align_up(off + (size_t)N * 4, 256);
    float* coef     = (float*)(ws + off);  off = align_up(off + (size_t)E * 4, 256);
    __bf16* xb      = (__bf16*)(ws + off); off = align_up(off + (size_t)N * DFEAT * 2, 256);
    __bf16* Wb      = (__bf16*)(ws + off); off = align_up(off + (size_t)DFEAT * DFEAT * 2, 256);
    float* y        = (float*)(ws + off);  off = align_up(off + (size_t)N * DFEAT * 4, 256);

    const int T = 256;

    // 1) degree (self-loops + edge weights) -> d^{-1/2}
    deg_init_kernel<<<(N + T - 1) / T, T, 0, stream>>>(node_deg, N);
    deg_accum_kernel<<<(E + T - 1) / T, T, 0, stream>>>(rows, ew, node_deg, E);
    dinv_kernel<<<(N + T - 1) / T, T, 0, stream>>>(node_deg, dinv, N);

    // 2) one-time fp32 -> bf16 conversion of x and W (memory bound)
    long long xn4 = (long long)N * DFEAT / 4;
    long long wn4 = (long long)DFEAT * DFEAT / 4;
    cvt_bf16_kernel<<<(unsigned)((xn4 + T - 1) / T), T, 0, stream>>>(x, xb, xn4);
    cvt_bf16_kernel<<<(unsigned)((wn4 + T - 1) / T), T, 0, stream>>>(W, Wb, wn4);

    // 3) y = x @ W^T + b via bf16 WMMA (f32 accumulate)
    const int mblocks = (N + 63) / 64;
    gemm_bias_wmma_kernel<<<mblocks, 256, 0, stream>>>(xb, Wb, b, y, N);

    // 4) aggregation (L2-resident gather + atomic scatter)
    coef_kernel<<<(E + T - 1) / T, T, 0, stream>>>(rows, cols, ew, dinv, coef, E);

    long long nf = (long long)N * DFEAT;
    selfloop_kernel<<<(unsigned)((nf + T - 1) / T), T, 0, stream>>>(y, dinv, out, N);

    long long ef = (long long)E * DFEAT;
    edge_scatter_kernel<<<(unsigned)((ef + T - 1) / T), T, 0, stream>>>(rows, cols, coef, y, out, E);

    // 5) ReLU
    relu_kernel<<<(unsigned)((nf + T - 1) / T), T, 0, stream>>>(out, nf);
}
